// HeteroGNNModel_91199335563383
// MI455X (gfx1250) — matrix operand order repaired and checked by reading, compiled, not run on hardware
//
#include <hip/hip_runtime.h>
#include <hip/hip_bf16.h>

// ---------------------------------------------------------------------------
// CDNA5 (gfx1250) HeteroGNN forward.
// Precision: node features kept in bf16 (halves gather traffic, feeds
// v_wmma_f32_16x16x32_bf16 directly); all accumulation in f32.
// Workspace layout (assumes ws_size >= 512MB):
//   [0)                bf16 xs_H/xs_C/xs_O   (1M nodes x 64 x 2B = 128MB)
//   [128MB)            f32  out_H/out_C/out_O (1M nodes x 64 x 4B = 256MB)
//   [384MB)            f32  msg buffer        (500K x 64 x 4B    = 128MB)
// ---------------------------------------------------------------------------

typedef __attribute__((ext_vector_type(16))) __bf16 v16bf;
typedef __attribute__((ext_vector_type(8)))  float  v8f;

__device__ __forceinline__ v8f wmma_bf16(v16bf a, v16bf b, v8f c) {
  // (neg_a, A, neg_b, B, c_mod, C, reuse_a, reuse_b)
  return __builtin_amdgcn_wmma_f32_16x16x32_bf16(false, a, false, b, (short)0, c,
                                                 false, false);
}

// ---- WMMA operand loaders (layouts per CDNA5 ISA 7.12.2) -------
// A matrix 16x32 (MxK), 16-bit:
//   lanes 0-15 : row M=lane,    elems K = [k0..k0+7] then [k0+16..k0+23]
//   lanes 16-31: row M=lane-16, elems K = [k0+8..k0+15] then [k0+24..k0+31]
__device__ __forceinline__ v16bf load_A_f32(const float* __restrict__ src, int ld,
                                            int row0, int k0, int lane) {
  int r  = row0 + (lane & 15);
  int kb = k0 + ((lane >> 4) << 3);
  const float* p = src + (size_t)r * ld + kb;
  v16bf a;
#pragma unroll
  for (int i = 0; i < 8; ++i) {
    a[i]     = (__bf16)p[i];
    a[i + 8] = (__bf16)p[i + 16];
  }
  return a;
}

__device__ __forceinline__ v16bf load_A_bf16(const __bf16* __restrict__ src, int ld,
                                             int row0, int k0, int lane) {
  int r  = row0 + (lane & 15);
  int kb = k0 + ((lane >> 4) << 3);
  const __bf16* p = src + (size_t)r * ld + kb;
  v16bf a;
#pragma unroll
  for (int i = 0; i < 8; ++i) {
    a[i]     = p[i];
    a[i + 8] = p[i + 16];
  }
  return a;
}

// B matrix 32x16 (KxN), 16-bit, W row-major KxN with leading dim ldn:
//   lanes 0-15 : col N=n0+lane,    K = [k0..k0+7] then [k0+16..k0+23]
//   lanes 16-31: col N=n0+lane-16, K = [k0+8..k0+15] then [k0+24..k0+31]
__device__ __forceinline__ v16bf load_B_f32(const float* __restrict__ W, int ldn,
                                            int k0, int n0, int lane) {
  int n  = n0 + (lane & 15);
  int kb = k0 + ((lane >> 4) << 3);
  v16bf b;
#pragma unroll
  for (int i = 0; i < 8; ++i) {
    b[i]     = (__bf16)W[(size_t)(kb + i) * ldn + n];
    b[i + 8] = (__bf16)W[(size_t)(kb + 16 + i) * ldn + n];
  }
  return b;
}

// C/D 16x16 f32: VGPR v -> lanes 0-15: (M=v, N=lane); lanes 16-31: (M=v+8, N=lane-16)
__device__ __forceinline__ v8f load_C(const float* __restrict__ dst, int ld,
                                      int row0, int n0, int lane) {
  int n  = n0 + (lane & 15);
  int mb = row0 + ((lane >> 4) << 3);
  v8f c;
#pragma unroll
  for (int v = 0; v < 8; ++v) c[v] = dst[(size_t)(mb + v) * ld + n];
  return c;
}

__device__ __forceinline__ void store_C(float* __restrict__ dst, int ld, int row0,
                                        int n0, const v8f c, int lane) {
  int n  = n0 + (lane & 15);
  int mb = row0 + ((lane >> 4) << 3);
#pragma unroll
  for (int v = 0; v < 8; ++v) dst[(size_t)(mb + v) * ld + n] = c[v];
}

// ---------------------------------------------------------------------------
// Encoder: xs = relu(relu(x @ W1 + b1) @ W2 + b2); x: Nt x 64 f32, out bf16.
// Block = 256 threads (8 waves), one block per 16 rows.
// ---------------------------------------------------------------------------
__global__ void k_encode(const float* __restrict__ x, const float* __restrict__ W1,
                         const float* __restrict__ b1, const float* __restrict__ W2,
                         const float* __restrict__ b2, __bf16* __restrict__ xs) {
  __shared__ __bf16 hid[16 * 128];
  const int lane = threadIdx.x & 31;
  const int wave = threadIdx.x >> 5;   // 0..7
  const int row0 = blockIdx.x * 16;

  // Stage 1: hidden tile, wave w -> hidden cols [16w, 16w+16)
  {
    const int n0 = wave * 16;
    v8f c = {};
#pragma unroll
    for (int k0 = 0; k0 < 64; k0 += 32) {
      v16bf a = load_A_f32(x, 64, row0, k0, lane);
      v16bf b = load_B_f32(W1, 128, k0, n0, lane);
      c = wmma_bf16(a, b, c);
    }
    const int n  = n0 + (lane & 15);
    const int mb = (lane >> 4) << 3;
    const float bias = b1[n];
#pragma unroll
    for (int v = 0; v < 8; ++v) {
      float t = c[v] + bias;
      t = t > 0.f ? t : 0.f;
      hid[(mb + v) * 128 + n] = (__bf16)t;
    }
  }
  __syncthreads();

  // Stage 2: out tile 16x64, waves 0..3 each take one 16-col tile, K=128
  if (wave < 4) {
    const int n0 = wave * 16;
    v8f c = {};
#pragma unroll
    for (int k0 = 0; k0 < 128; k0 += 32) {
      v16bf a = load_A_bf16(hid, 128, 0, k0, lane);
      v16bf b = load_B_f32(W2, 64, k0, n0, lane);
      c = wmma_bf16(a, b, c);
    }
    const int n  = n0 + (lane & 15);
    const int mb = (lane >> 4) << 3;
    const float bias = b2[n];
#pragma unroll
    for (int v = 0; v < 8; ++v) {
      float t = c[v] + bias;
      t = t > 0.f ? t : 0.f;
      xs[(size_t)(row0 + mb + v) * 64 + n] = (__bf16)t;
    }
  }
}

// ---------------------------------------------------------------------------
// out[i][j] = br[r0][j] + br[r1][j] + br[r2][j]  (relations with dst type d
// are r = d, d+3, d+6 in the RELS ordering)
// ---------------------------------------------------------------------------
__global__ void k_bias_init(float* __restrict__ out, const float* __restrict__ br_l,
                            int d, long long n64) {
  long long i = (long long)blockIdx.x * blockDim.x + threadIdx.x;
  if (i >= n64) return;
  int j = (int)(i & 63);
  out[i] = br_l[(d + 0) * 64 + j] + br_l[(d + 3) * 64 + j] + br_l[(d + 6) * 64 + j];
}

__global__ void k_zero(float* __restrict__ p, long long n) {
  long long i = (long long)blockIdx.x * blockDim.x + threadIdx.x;
  if (i < n) p[i] = 0.f;
}

// ---------------------------------------------------------------------------
// Edge scatter-sum: msg[dst[e]] += xs_src[src[e]]  (8 threads per edge,
// 8 floats each; f32 global atomics -> global_atomic_add_f32)
// ---------------------------------------------------------------------------
__global__ void k_scatter(const int* __restrict__ src, const int* __restrict__ dst,
                          const __bf16* __restrict__ xs_s, float* __restrict__ msg,
                          int nedges) {
  long long tid = (long long)blockIdx.x * blockDim.x + threadIdx.x;
  int e  = (int)(tid >> 3);
  int c8 = ((int)tid & 7) * 8;
  if (e >= nedges) return;
  const int s = src[e];
  const int d = dst[e];
  const __bf16* p = xs_s + (size_t)s * 64 + c8;
  float*         q = msg  + (size_t)d * 64 + c8;
#pragma unroll
  for (int i = 0; i < 8; ++i) atomicAdd(q + i, (float)p[i]);
}

// ---------------------------------------------------------------------------
// Per-relation GEMM: out += bf16(msg) @ Wr + xs_d @ Wroot
// Block = 128 threads (4 waves), one block per 16 rows; wave w -> col tile w.
// ---------------------------------------------------------------------------
__global__ void k_gemm_rel(const float* __restrict__ msg, const __bf16* __restrict__ xs_d,
                           const float* __restrict__ Wr, const float* __restrict__ Wroot,
                           float* __restrict__ out) {
  const int lane = threadIdx.x & 31;
  const int wave = threadIdx.x >> 5;   // 0..3
  const int row0 = blockIdx.x * 16;
  const int n0   = wave * 16;
  v8f c = load_C(out, 64, row0, n0, lane);
#pragma unroll
  for (int k0 = 0; k0 < 64; k0 += 32) {
    v16bf am = load_A_f32(msg, 64, row0, k0, lane);
    v16bf bm = load_B_f32(Wr, 64, k0, n0, lane);
    c = wmma_bf16(am, bm, c);
    v16bf ar = load_A_bf16(xs_d, 64, row0, k0, lane);
    v16bf br = load_B_f32(Wroot, 64, k0, n0, lane);
    c = wmma_bf16(ar, br, c);
  }
  store_C(out, 64, row0, n0, c, lane);
}

// xs = bf16(relu(out))
__global__ void k_relu_cvt(const float* __restrict__ out, __bf16* __restrict__ xs,
                           long long n) {
  long long i = (long long)blockIdx.x * blockDim.x + threadIdx.x;
  if (i >= n) return;
  float t = out[i];
  xs[i] = (__bf16)(t > 0.f ? t : 0.f);
}

// head: o[i] = xs[i,:] . W + b  (D=64 -> 1)
__global__ void k_head(const __bf16* __restrict__ xs, const float* __restrict__ W,
                       const float* __restrict__ b, float* __restrict__ o, int Nt) {
  int i = blockIdx.x * blockDim.x + threadIdx.x;
  if (i >= Nt) return;
  const __bf16* p = xs + (size_t)i * 64;
  float acc = b[0];
#pragma unroll
  for (int j = 0; j < 64; ++j) acc += (float)p[j] * W[j];
  o[i] = acc;
}

// ---------------------------------------------------------------------------
extern "C" void kernel_launch(void* const* d_in, const int* in_sizes, int n_in,
                              void* d_out, int out_size, void* d_ws, size_t ws_size,
                              hipStream_t stream) {
  (void)in_sizes; (void)n_in; (void)out_size; (void)ws_size;

  static const int NT[3] = {500000, 400000, 100000};
  const int E = 1000000;
  const int L = 2;

  const float* x[3]  = {(const float*)d_in[0], (const float*)d_in[1], (const float*)d_in[2]};
  const int*   ed[9];
  for (int r = 0; r < 9; ++r) ed[r] = (const int*)d_in[3 + r];
  const float* enc_W1 = (const float*)d_in[12];  // (3,64,128)
  const float* enc_b1 = (const float*)d_in[13];  // (3,128)
  const float* enc_W2 = (const float*)d_in[14];  // (3,128,64)
  const float* enc_b2 = (const float*)d_in[15];  // (3,64)
  const float* rel_Wr    = (const float*)d_in[16]; // (2,9,64,64)
  const float* rel_br    = (const float*)d_in[17]; // (2,9,64)
  const float* rel_Wroot = (const float*)d_in[18]; // (2,9,64,64)
  const float* head_W = (const float*)d_in[19];    // (2,64,1)
  const float* head_b = (const float*)d_in[20];    // (2,1)

  // workspace carve-up
  char* ws = (char*)d_ws;
  size_t off = 0;
  __bf16* xs[3];
  for (int t = 0; t < 3; ++t) { xs[t] = (__bf16*)(ws + off); off += (size_t)NT[t] * 64 * sizeof(__bf16); }
  float* acc[3];
  for (int t = 0; t < 3; ++t) { acc[t] = (float*)(ws + off); off += (size_t)NT[t] * 64 * sizeof(float); }
  float* msg = (float*)(ws + off);  // max 500000*64 f32

  // ---- encoders ----
  for (int t = 0; t < 3; ++t) {
    k_encode<<<NT[t] / 16, 256, 0, stream>>>(
        x[t], enc_W1 + (size_t)t * 64 * 128, enc_b1 + (size_t)t * 128,
        enc_W2 + (size_t)t * 128 * 64, enc_b2 + (size_t)t * 64, xs[t]);
  }

  // ---- GNN layers ----
  for (int l = 0; l < L; ++l) {
    const float* br_l = rel_br + (size_t)l * 9 * 64;
    for (int d = 0; d < 3; ++d) {
      long long n64 = (long long)NT[d] * 64;
      k_bias_init<<<(unsigned)((n64 + 255) / 256), 256, 0, stream>>>(acc[d], br_l, d, n64);
    }
    for (int r = 0; r < 9; ++r) {
      const int s = r / 3, d = r % 3;
      long long nm = (long long)NT[d] * 64;
      k_zero<<<(unsigned)((nm + 255) / 256), 256, 0, stream>>>(msg, nm);
      long long nth = (long long)E * 8;
      k_scatter<<<(unsigned)((nth + 255) / 256), 256, 0, stream>>>(
          ed[r], ed[r] + E, xs[s], msg, E);
      k_gemm_rel<<<NT[d] / 16, 128, 0, stream>>>(
          msg, xs[d], rel_Wr + ((size_t)l * 9 + r) * 64 * 64,
          rel_Wroot + ((size_t)l * 9 + r) * 64 * 64, acc[d]);
    }
    for (int d = 0; d < 3; ++d) {
      long long n64 = (long long)NT[d] * 64;
      k_relu_cvt<<<(unsigned)((n64 + 255) / 256), 256, 0, stream>>>(acc[d], xs[d], n64);
    }
  }

  // ---- heads (H then C, concatenated into d_out) ----
  float* out = (float*)d_out;
  k_head<<<(NT[0] + 255) / 256, 256, 0, stream>>>(xs[0], head_W, head_b, out, NT[0]);
  k_head<<<(NT[1] + 255) / 256, 256, 0, stream>>>(xs[1], head_W + 64, head_b + 1,
                                                  out + NT[0], NT[1]);
}
